// TriangleAttention_68384469286924
// MI455X (gfx1250) — matrix operand order repaired and checked by reading, not compiled, not run on hardware
//
#include <hip/hip_runtime.h>

#define Dm 128
#define Hh 4
#define Nn 320
#define Pp (Nn*Nn)

typedef unsigned short u16;
typedef unsigned int u32;

typedef __attribute__((ext_vector_type(16))) __bf16 v16bf;
typedef __attribute__((ext_vector_type(16))) unsigned short v16u;
typedef __attribute__((ext_vector_type(8))) float v8f;

__device__ __forceinline__ u16 f2bf(float f){
  u32 u = __float_as_uint(f);
  u32 r = u + 0x7FFFu + ((u >> 16) & 1u);
  return (u16)(r >> 16);
}
__device__ __forceinline__ float bf2f(u16 h){ return __uint_as_float(((u32)h) << 16); }

__device__ __forceinline__ v8f wmma_bf16(v16u a, v16u b, v8f c){
  return __builtin_amdgcn_wmma_f32_16x16x32_bf16(
      false, __builtin_bit_cast(v16bf, a),
      false, __builtin_bit_cast(v16bf, b),
      (short)0, c, false, false);
}

union FragU { v16u v; uint4 q[2]; };

// A-fragment 16x32 bf16 from LDS tile, row-major rows x K with leading dim ld.
// Lane (lo,hi): element e <-> K = kbase+hi*16+e
__device__ __forceinline__ v16u lds_afrag(const u16* tile, int ld, int kbase, int lane){
  int lo = lane & 15, hi = lane >> 4;
  const u16* row = tile + lo*ld + kbase;
  FragU u;
  u.q[0] = *(const uint4*)(row + hi*16);
  u.q[1] = *(const uint4*)(row + hi*16 + 8);
  return u.v;
}

// B-fragment 32x16: source laid out [n][k] with K contiguous, stride ldk.
// Lane (n,hi): element e <-> K = kbase + hi*16 + e  (one 32B contiguous run)
__device__ __forceinline__ v16u lds_bfrag(const u16* wt, int ldk, int kbase, int lane){
  int n = lane & 15, hi = lane >> 4;
  return *(const v16u*)(wt + n*ldk + kbase + hi*16);
}

__device__ __forceinline__ float red_sum32(float v){
#pragma unroll
  for(int m=16;m>=1;m>>=1) v += __shfl_xor(v, m, 32);
  return v;
}
__device__ __forceinline__ float red_max16(float v){
#pragma unroll
  for(int m=8;m>=1;m>>=1) v = fmaxf(v, __shfl_xor(v, m, 32));
  return v;
}
__device__ __forceinline__ float red_sum16(float v){
#pragma unroll
  for(int m=8;m>=1;m>>=1) v += __shfl_xor(v, m, 32);
  return v;
}

// ---------------- Kernel 0: convert weights to bf16 (Wcat = [128][Q|K|V|G], Wo) -----
__global__ void k_wconv(const float* __restrict__ Wq, const float* __restrict__ Wk,
                        const float* __restrict__ Wv, const float* __restrict__ Wg,
                        const float* __restrict__ Wo,
                        u16* __restrict__ Wcat, u16* __restrict__ Wobf){
  int idx = blockIdx.x*256 + threadIdx.x;
  if (idx < 128*512){
    int k = idx >> 9, n = idx & 511;
    const float* src = (n < 128) ? Wq : (n < 256) ? Wk : (n < 384) ? Wv : Wg;
    Wcat[idx] = f2bf(src[k*128 + (n & 127)]);
  } else if (idx < 128*512 + 128*128){
    int t = idx - 128*512;
    Wobf[t] = f2bf(Wo[t]);
  }
}

// ---------------- Kernel 1: LayerNorm1 -> z (bf16) ; B = z @ Wb (f32) ---------------
__global__ void __launch_bounds__(256) k_ln_in(const float* __restrict__ pair,
                        const float* __restrict__ g1, const float* __restrict__ b1,
                        const float* __restrict__ Wb,
                        u16* __restrict__ Zb, float* __restrict__ Bb){
  int wid = threadIdx.x >> 5, lane = threadIdx.x & 31;
  int pos = blockIdx.x*8 + wid;
  float4 x = ((const float4*)(pair + (size_t)pos*Dm))[lane];
  float s = red_sum32(x.x + x.y + x.z + x.w);
  float mean = s * (1.0f/128.0f);
  float d0 = x.x-mean, d1 = x.y-mean, d2 = x.z-mean, d3 = x.w-mean;
  float ss = red_sum32(d0*d0 + d1*d1 + d2*d2 + d3*d3);
  float rstd = rsqrtf(ss*(1.0f/128.0f) + 1e-5f);
  float4 g = ((const float4*)g1)[lane];
  float4 b = ((const float4*)b1)[lane];
  float z0 = d0*rstd*g.x + b.x;
  float z1 = d1*rstd*g.y + b.y;
  float z2 = d2*rstd*g.z + b.z;
  float z3 = d3*rstd*g.w + b.w;
  uint2 o;
  o.x = (u32)f2bf(z0) | ((u32)f2bf(z1) << 16);
  o.y = (u32)f2bf(z2) | ((u32)f2bf(z3) << 16);
  ((uint2*)(Zb + (size_t)pos*Dm))[lane] = o;
  // per-key bias: B[pos][h] = sum_k z[k] * Wb[k][h]
  float zz[4] = {z0, z1, z2, z3};
  float a0=0.f, a1=0.f, a2=0.f, a3=0.f;
#pragma unroll
  for(int c=0;c<4;++c){
    float4 w = ((const float4*)Wb)[lane*4 + c];
    a0 += zz[c]*w.x; a1 += zz[c]*w.y; a2 += zz[c]*w.z; a3 += zz[c]*w.w;
  }
  a0 = red_sum32(a0); a1 = red_sum32(a1); a2 = red_sum32(a2); a3 = red_sum32(a3);
  if (lane == 0){
    float4 o4; o4.x=a0; o4.y=a1; o4.z=a2; o4.w=a3;
    ((float4*)Bb)[pos] = o4;
  }
}

// ---------------- Kernel 2: fused QKVG GEMM: z[102400,128] x Wcat[128,512] ---------
__global__ void __launch_bounds__(256) k_qkvg(const u16* __restrict__ Zb, const u16* __restrict__ Wcat,
                       const float* __restrict__ bg,
                       u16* __restrict__ Qb, u16* __restrict__ Kb,
                       u16* __restrict__ Vb, u16* __restrict__ Gb){
  __shared__ __align__(32) u16 lZ[32*128];     // 8 KB  (rows x K)
  __shared__ __align__(32) u16 lWt[128*128];   // 32 KB (n x K, transposed)
  int tid = threadIdx.x;
  int mb = blockIdx.x;        // 32-row tile
  int cb = blockIdx.y * 128;  // 128-col block of 512
  {
    const u32* zsrc = (const u32*)(Zb + (size_t)mb*32*Dm);
    u32* zdst = (u32*)lZ;
#pragma unroll
    for(int q=0;q<8;++q) zdst[tid + q*256] = zsrc[tid + q*256];
    for(int q=0;q<64;++q){
      int idx = tid + q*256;
      int n = idx >> 7, k = idx & 127;
      lWt[n*128 + k] = Wcat[k*512 + cb + n];
    }
  }
  __syncthreads();
  int wid = tid >> 5, lane = tid & 31;
  int mhalf = wid >> 2, ngroup = wid & 3;
  v8f acc0 = {0,0,0,0,0,0,0,0};
  v8f acc1 = acc0;
#pragma unroll
  for(int kc=0;kc<4;++kc){
    v16u a  = lds_afrag(lZ + mhalf*16*128, 128, kc*32, lane);
    v16u b0 = lds_bfrag(lWt + (ngroup*32     )*128, 128, kc*32, lane);
    v16u b1 = lds_bfrag(lWt + (ngroup*32 + 16)*128, 128, kc*32, lane);
    acc0 = wmma_bf16(a, b0, acc0);
    acc1 = wmma_bf16(a, b1, acc1);
  }
  int lo = lane & 15, hi = lane >> 4;
#pragma unroll
  for(int nt=0;nt<2;++nt){
    v8f acc = nt ? acc1 : acc0;
    int c = cb + ngroup*32 + nt*16 + lo;   // 0..511 (tile category is uniform)
#pragma unroll
    for(int r=0;r<8;++r){
      int gm = mb*32 + mhalf*16 + r + 8*hi;
      float v = acc[r];
      if (c < 128)       Qb[(size_t)gm*Dm + c]         = f2bf(v * 0.17677669529663687f);
      else if (c < 256)  Kb[(size_t)gm*Dm + (c - 128)] = f2bf(v);
      else if (c < 384)  Vb[(size_t)gm*Dm + (c - 256)] = f2bf(v);
      else               Gb[(size_t)gm*Dm + (c - 384)] =
                           f2bf(1.0f / (1.0f + __expf(-(v + bg[c - 384]))));
    }
  }
}

// ---------------- Kernel 3: flash attention per (l, head, 64-row i-block) ----------
__global__ void __launch_bounds__(128) k_attn(const u16* __restrict__ Qb, const u16* __restrict__ Kb,
                      const u16* __restrict__ Vb, const u16* __restrict__ Gb,
                      const float* __restrict__ Bb, u16* __restrict__ GO){
  __shared__ __align__(32) u16 lQ [64*32];    // 4 KB   (i x k)
  __shared__ __align__(32) u16 lK [320*32];   // 20 KB  (j x k)  -> K^T B-frags contiguous
  __shared__ __align__(32) u16 lVt[32*320];   // 20 KB  (d x j)  -> V B-frags contiguous
  __shared__ __align__(32) u16 lP [4*16*32];  // 4 KB per-wave P scratch
  int tid = threadIdx.x;
  int ib = blockIdx.x, h = blockIdx.y, l = blockIdx.z;
  size_t base = (size_t)l * Nn;
  int hc = h * 32;
  {
    u32* qd = (u32*)lQ;
#pragma unroll
    for(int q=0;q<8;++q){
      int idx = tid + q*128;                 // 1024 dwords
      int r = idx >> 4, cu = idx & 15;
      qd[idx] = ((const u32*)(Qb + (base + ib*64 + r)*Dm + hc))[cu];
    }
    u32* kd = (u32*)lK;
    for(int q=0;q<40;++q){
      int idx = tid + q*128;                 // 5120 dwords
      int j = idx >> 4, cu = idx & 15;
      kd[idx] = ((const u32*)(Kb + (base + j)*Dm + hc))[cu];
    }
    for(int q=0;q<40;++q){                   // V transposed: lVt[d][j]
      int idx = tid + q*128;                 // 5120 dword-pairs
      int j = idx >> 4, d2 = (idx & 15)*2;
      u32 v = ((const u32*)(Vb + (base + j)*Dm + hc))[idx & 15];
      lVt[d2*320 + j]     = (u16)(v & 0xFFFFu);
      lVt[(d2+1)*320 + j] = (u16)(v >> 16);
    }
  }
  __syncthreads();
  int wid = tid >> 5, lane = tid & 31, lo = lane & 15, hi = lane >> 4;
  u16* lPw = lP + wid*16*32;
  v16u qa = lds_afrag(lQ + wid*16*32, 32, 0, lane);   // loop-invariant Q fragment
  v8f z8 = {0,0,0,0,0,0,0,0};
  v8f acc0 = z8, acc1 = z8;
  float mrow[8], lrow[8];
#pragma unroll
  for(int r=0;r<8;++r){ mrow[r] = -1e30f; lrow[r] = 0.f; }
  for(int jc=0;jc<10;++jc){
    int j0 = jc*32, j1 = j0 + 16;
    v16u kb0 = lds_bfrag(lK + j0*32, 32, 0, lane);
    v16u kb1 = lds_bfrag(lK + j1*32, 32, 0, lane);
    v8f s0 = wmma_bf16(qa, kb0, z8);
    v8f s1 = wmma_bf16(qa, kb1, z8);
    float bias0 = Bb[(base + j0 + lo)*4 + h];
    float bias1 = Bb[(base + j1 + lo)*4 + h];
#pragma unroll
    for(int r=0;r<8;++r){
      float a = s0[r] + bias0;
      float b = s1[r] + bias1;
      float t  = red_max16(fmaxf(a, b));
      float mn = fmaxf(mrow[r], t);
      float sc = __expf(mrow[r] - mn);
      float p0 = __expf(a - mn), p1 = __expf(b - mn);
      float rs = red_sum16(p0 + p1);
      lrow[r] = lrow[r]*sc + rs;
      mrow[r] = mn;
      acc0[r] *= sc; acc1[r] *= sc;
      int row = r + 8*hi;
      lPw[row*32 + lo]      = f2bf(p0);
      lPw[row*32 + 16 + lo] = f2bf(p1);
    }
    v16u pa  = lds_afrag(lPw, 32, 0, lane);            // per-wave, LDS ops in-order
    v16u vb0 = lds_bfrag(lVt,          320, j0, lane); // d = 0..15
    v16u vb1 = lds_bfrag(lVt + 16*320, 320, j0, lane); // d = 16..31
    acc0 = wmma_bf16(pa, vb0, acc0);
    acc1 = wmma_bf16(pa, vb1, acc1);
  }
#pragma unroll
  for(int r=0;r<8;++r){
    float inv = 1.0f / lrow[r];
    int i = ib*64 + wid*16 + r + 8*hi;
    size_t pos = base + i;
    int c0 = hc + lo, c1 = c0 + 16;
    float o0 = acc0[r]*inv * bf2f(Gb[pos*Dm + c0]);
    float o1 = acc1[r]*inv * bf2f(Gb[pos*Dm + c1]);
    GO[pos*Dm + c0] = f2bf(o0);
    GO[pos*Dm + c1] = f2bf(o1);
  }
}

// ---------------- Kernel 4: out projection + residual + LayerNorm2 -----------------
__global__ void __launch_bounds__(256) k_out(const u16* __restrict__ GO, const u16* __restrict__ Wobf,
                     const float* __restrict__ bo, const float* __restrict__ pair,
                     const float* __restrict__ g2, const float* __restrict__ b2,
                     float* __restrict__ out){
  __shared__ __align__(32) u16 lGO[32*128];   // 8 KB
  __shared__ __align__(32) u16 lWt[128*128];  // 32 KB (n x k)
  __shared__ float lX[32*128];                // 16 KB
  __shared__ float lm[32], lv[32];
  int tid = threadIdx.x;
  int mb = blockIdx.x;
  {
    const u32* gsrc = (const u32*)(GO + (size_t)mb*32*Dm);
    u32* gdst = (u32*)lGO;
#pragma unroll
    for(int q=0;q<8;++q) gdst[tid + q*256] = gsrc[tid + q*256];
    for(int q=0;q<64;++q){
      int idx = tid + q*256;
      int n = idx >> 7, k = idx & 127;
      lWt[n*128 + k] = Wobf[k*128 + n];
    }
  }
  __syncthreads();
  int wid = tid >> 5, lane = tid & 31, lo = lane & 15, hi = lane >> 4;
  int mhalf = wid >> 2, ngroup = wid & 3;
  v8f acc0 = {0,0,0,0,0,0,0,0};
  v8f acc1 = acc0;
#pragma unroll
  for(int kc=0;kc<4;++kc){
    v16u a  = lds_afrag(lGO + mhalf*16*128, 128, kc*32, lane);
    v16u b0 = lds_bfrag(lWt + (ngroup*32     )*128, 128, kc*32, lane);
    v16u b1 = lds_bfrag(lWt + (ngroup*32 + 16)*128, 128, kc*32, lane);
    acc0 = wmma_bf16(a, b0, acc0);
    acc1 = wmma_bf16(a, b1, acc1);
  }
#pragma unroll
  for(int nt=0;nt<2;++nt){
    v8f acc = nt ? acc1 : acc0;
    int c = ngroup*32 + nt*16 + lo;
#pragma unroll
    for(int r=0;r<8;++r){
      int rowl = mhalf*16 + r + 8*hi;
      size_t gm = (size_t)mb*32 + rowl;
      lX[rowl*128 + c] = acc[r] + bo[c] + pair[gm*Dm + c];
    }
  }
  __syncthreads();
  if (tid < 32){
    const float* xr = lX + tid*128;
    float s = 0.f;
    for(int c=0;c<128;++c) s += xr[c];
    float m = s * (1.0f/128.0f);
    float ss = 0.f;
    for(int c=0;c<128;++c){ float d = xr[c] - m; ss += d*d; }
    lm[tid] = m;
    lv[tid] = rsqrtf(ss*(1.0f/128.0f) + 1e-5f);
  }
  __syncthreads();
#pragma unroll
  for(int q=0;q<16;++q){
    int e = tid + q*256;
    int rowl = e >> 7, c = e & 127;
    size_t gm = (size_t)mb*32 + rowl;
    out[gm*Dm + c] = (lX[e] - lm[rowl]) * lv[rowl] * g2[c] + b2[c];
  }
}

extern "C" void kernel_launch(void* const* d_in, const int* in_sizes, int n_in,
                              void* d_out, int out_size, void* d_ws, size_t ws_size,
                              hipStream_t stream){
  (void)in_sizes; (void)n_in; (void)out_size; (void)ws_size;
  const float* pair = (const float*)d_in[0];
  const float* g1   = (const float*)d_in[1];
  const float* b1   = (const float*)d_in[2];
  const float* Wq   = (const float*)d_in[3];
  const float* Wk   = (const float*)d_in[4];
  const float* Wv   = (const float*)d_in[5];
  const float* Wb   = (const float*)d_in[6];
  const float* Wg   = (const float*)d_in[7];
  const float* bg   = (const float*)d_in[8];
  const float* Wo   = (const float*)d_in[9];
  const float* bo   = (const float*)d_in[10];
  const float* g2   = (const float*)d_in[11];
  const float* b2   = (const float*)d_in[12];
  float* out = (float*)d_out;

  char* w = (char*)d_ws;
  const size_t ZBYTES = (size_t)Pp * Dm * 2;   // 26,214,400 bytes per bf16 plane
  u16* Zb   = (u16*)(w);
  u16* Qb   = (u16*)(w + 1*ZBYTES);
  u16* Kb   = (u16*)(w + 2*ZBYTES);
  u16* Vb   = (u16*)(w + 3*ZBYTES);
  u16* Gb   = (u16*)(w + 4*ZBYTES);
  u16* GO   = (u16*)(w + 5*ZBYTES);
  float* Bb = (float*)(w + 6*ZBYTES);                                  // Pp*4 floats
  u16* Wcat = (u16*)(w + 6*ZBYTES + (size_t)Pp*16);                    // 128*512 bf16
  u16* Wobf = (u16*)(w + 6*ZBYTES + (size_t)Pp*16 + 128*512*2);        // 128*128 bf16

  k_wconv<<<(128*512 + 128*128 + 255)/256, 256, 0, stream>>>(Wq, Wk, Wv, Wg, Wo, Wcat, Wobf);
  k_ln_in<<<Pp/8, 256, 0, stream>>>(pair, g1, b1, Wb, Zb, Bb);
  k_qkvg<<<dim3(Pp/32, 4), 256, 0, stream>>>(Zb, Wcat, bg, Qb, Kb, Vb, Gb);
  k_attn<<<dim3(Nn/64, Hh, Nn), 128, 0, stream>>>(Qb, Kb, Vb, Gb, Bb, GO);
  k_out<<<Pp/32, 256, 0, stream>>>(GO, Wobf, bo, pair, g2, b2, out);
}
